// WaveNet_9792525435573
// MI455X (gfx1250) — matrix operand order repaired and checked by reading, not compiled, not run on hardware
//
#include <hip/hip_runtime.h>
#include <hip/hip_bf16.h>
#include <math.h>

typedef __attribute__((ext_vector_type(16))) _Float16 v16h;
typedef __attribute__((ext_vector_type(8)))  _Float16 v8h;
typedef __attribute__((ext_vector_type(8)))  float    v8f;

union Frag { v16h v; _Float16 h[16]; v8h p[2]; };
union Acc  { v8f  v; float f[8]; };

#define B_      4
#define T_WAV   10240
#define T2_     42
#define N_LC_IN 80
#define N_SPK   40
#define NLAY    20
#define TOTAL_D 2046
#define T_OUT   (T_WAV - TOTAL_D)   /* 8194 */
#define TPAD    (T_WAV + 96)
#define TOPAD   (T_OUT + 96)
#define T_U3    12375
#define LC_OFF  1067

__device__ __forceinline__ v8f wmma16(v16h a, v16h b, v8f c) {
  return __builtin_amdgcn_wmma_f32_16x16x32_f16(false, a, false, b, (short)0, c, false, false);
}

// A-fragment (16x32 f16): lane<16 row M=lane, elems 0..7 = K k0+half*8.., elems 8..15 = K k0+16+half*8..
__device__ __forceinline__ v16h load_fragA(const _Float16* __restrict__ p, int row0, int ld, int k0) {
  int lane = threadIdx.x & 31;
  int half = lane >> 4, r = lane & 15;
  const _Float16* q = p + (size_t)(row0 + r) * ld + k0 + half * 8;
  Frag f;
  f.p[0] = *(const v8h*)(q);
  f.p[1] = *(const v8h*)(q + 16);
  return f.v;
}

// B-fragment (32x16 f16), weights stored (N,K) row-major: col N=lane&15, elems 0..7 = K k0+half*8..,
// elems 8..15 = K k0+16+half*8.. (same order as A-fragment).
__device__ __forceinline__ v16h load_fragB(const _Float16* __restrict__ w, int n0, int ld, int k0) {
  int lane = threadIdx.x & 31;
  int half = lane >> 4, n = lane & 15;
  const _Float16* q = w + (size_t)(n0 + n) * ld + k0 + half * 8;
  Frag f;
  f.p[0] = *(const v8h*)(q);
  f.p[1] = *(const v8h*)(q + 16);
  return f.v;
}

// A-fragment from a [16 x ld] f16 LDS tile
__device__ __forceinline__ v16h load_fragA_lds(const _Float16* tile, int ld, int k0) {
  int lane = threadIdx.x & 31;
  int half = lane >> 4, r = lane & 15;
  const _Float16* q = tile + r * ld + k0 + half * 8;
  Frag f;
  f.p[0] = *(const v8h*)(q);
  f.p[1] = *(const v8h*)(q + 16);
  return f.v;
}

// ---------------- simple prep kernels ----------------

__global__ void k_zero_f32(float* p, long n) {
  long i = (long)blockIdx.x * blockDim.x + threadIdx.x;
  if (i < n) p[i] = 0.f;
}

// (L,O,I,taps) f32 -> (L,taps,O,I) f16
__global__ void k_w2t(const float* __restrict__ src, _Float16* __restrict__ dst,
                      int L, int O, int I, int taps) {
  long idx = (long)blockIdx.x * blockDim.x + threadIdx.x;
  long total = (long)L * O * I * taps;
  if (idx >= total) return;
  long per = (long)O * I * taps;
  int l = (int)(idx / per);
  long rem = idx % per;
  int tap = (int)(rem / ((long)O * I));
  long rem2 = rem % ((long)O * I);
  int o = (int)(rem2 / I), i = (int)(rem2 % I);
  dst[idx] = (_Float16)src[(size_t)l * per + ((size_t)o * I + i) * taps + tap];
}

// wav (B,256,T) f32 -> (B,TPAD,256) f16
__global__ void k_wavT(const float* __restrict__ wav, _Float16* __restrict__ out) {
  long idx = (long)blockIdx.x * blockDim.x + threadIdx.x;
  long total = (long)B_ * T_WAV * 256;
  if (idx >= total) return;
  int b = (int)(idx / ((long)T_WAV * 256));
  int t = (int)((idx / 256) % T_WAV);
  int c = (int)(idx % 256);
  out[((size_t)b * TPAD + t) * 256 + c] = (_Float16)wav[((size_t)b * 256 + c) * T_WAV + t];
}

// jitter gather + k=3 valid conv: out (B,128,40) f32
__global__ void k_lcconv(const float* __restrict__ lc, const int* __restrict__ jit,
                         const float* __restrict__ W, const float* __restrict__ bias,
                         float* __restrict__ out) {
  int idx = blockIdx.x * blockDim.x + threadIdx.x;
  int total = B_ * 128 * 40;
  if (idx >= total) return;
  int b = idx / (128 * 40);
  int o = (idx / 40) % 128;
  int t = idx % 40;
  float s = bias[o];
  for (int k = 0; k < 3; k++) {
    int jt = jit[b * T2_ + t + k];
    const float* xp = lc + ((size_t)b * N_LC_IN) * T2_ + jt;
    const float* wp = W + ((size_t)o * N_LC_IN) * 3 + k;
    for (int i = 0; i < N_LC_IN; i++)
      s += wp[(size_t)i * 3] * xp[(size_t)i * T2_];
  }
  out[idx] = s;
}

// transposed conv per reference: out[o,t] = b[o] + sum_{i,j} w[i][o][j]*x[i][(t+f-s-j)/s]
__global__ void k_upconv(const float* __restrict__ x, const float* __restrict__ w,
                         const float* __restrict__ bias, float* __restrict__ out,
                         int Tin, int Tout, int s, int f) {
  long idx = (long)blockIdx.x * blockDim.x + threadIdx.x;
  long total = (long)B_ * 128 * Tout;
  if (idx >= total) return;
  int b = (int)(idx / ((long)128 * Tout));
  int o = (int)((idx / Tout) % 128);
  int t = (int)(idx % Tout);
  float acc = bias[o];
  for (int j = 0; j < f; j++) {
    int num = t + f - s - j;
    if (num < 0 || (num % s) != 0) continue;
    int ti = num / s;
    if (ti >= Tin) continue;
    const float* xp = x + ((size_t)b * 128) * Tin + ti;
    const float* wp = w + (size_t)o * f + j;
    for (int i = 0; i < 128; i++)
      acc += wp[(size_t)i * 128 * f] * xp[(size_t)i * Tin];
  }
  out[idx] = acc;
}

// cond (B,TPAD,256) f16: [0:128]=lc crop, [128:256]=speaker embedding
__global__ void k_cond(const float* __restrict__ lcf, const float* __restrict__ W_spk,
                       const float* __restrict__ b_spk, const int* __restrict__ spk,
                       _Float16* __restrict__ cond) {
  long idx = (long)blockIdx.x * blockDim.x + threadIdx.x;
  long total = (long)B_ * T_WAV * 256;
  if (idx >= total) return;
  int b = (int)(idx / ((long)T_WAV * 256));
  int t = (int)((idx / 256) % T_WAV);
  int c = (int)(idx % 256);
  float v;
  if (c < 128) v = lcf[((size_t)b * 128 + c) * T_U3 + (LC_OFF + t)];
  else         v = W_spk[(size_t)(c - 128) * N_SPK + spk[b]] + b_spk[c - 128];
  cond[((size_t)b * TPAD + t) * 256 + c] = (_Float16)v;
}

// ---------------- WMMA GEMM kernels ----------------

// out(B,TPAD,128) f16 = X(B,TPAD,K) f16 @ Wt(128,K)^T + bias
__global__ __launch_bounds__(128) void k_gemm_n128(const _Float16* __restrict__ X,
                                                   const _Float16* __restrict__ Wt,
                                                   const float* __restrict__ bias,
                                                   _Float16* __restrict__ out,
                                                   int rows, int K) {
  int b = blockIdx.y;
  int wave = threadIdx.x >> 5, lane = threadIdx.x & 31;
  int half = lane >> 4, lc = lane & 15;
  int t0 = blockIdx.x * 64 + wave * 16;
  __shared__ _Float16 st[4][16 * 128];
  const _Float16* Xb = X + (size_t)b * TPAD * K;
  Acc acc[8];
#pragma unroll
  for (int nt = 0; nt < 8; nt++) {
    float v = bias[nt * 16 + lc];
    for (int r = 0; r < 8; r++) acc[nt].f[r] = v;
  }
  for (int kt = 0; kt < K / 32; kt++) {
    v16h a = load_fragA(Xb, t0, K, kt * 32);
#pragma unroll
    for (int nt = 0; nt < 8; nt++)
      acc[nt].v = wmma16(a, load_fragB(Wt, nt * 16, K, kt * 32), acc[nt].v);
  }
  _Float16* tile = st[wave];
#pragma unroll
  for (int nt = 0; nt < 8; nt++) {
    int n = nt * 16 + lc;
    for (int r = 0; r < 8; r++) tile[(half * 8 + r) * 128 + n] = (_Float16)acc[nt].f[r];
  }
  __syncthreads();
  int r = lane & 15, hh = lane >> 4;
  int tl = t0 + r;
  if (tl < rows) {
    const v8h* src = (const v8h*)(tile + r * 128 + hh * 64);
    v8h* dst = (v8h*)(out + ((size_t)b * TPAD + tl) * 128 + hh * 64);
#pragma unroll
    for (int i = 0; i < 8; i++) dst[i] = src[i];
  }
}

// one WaveNet layer (fused dilated conv + cond + gate + skip + res)
__global__ __launch_bounds__(128) void k_layer(const _Float16* __restrict__ sig_in,
                                               _Float16* __restrict__ sig_out,
                                               const _Float16* __restrict__ cond,
                                               float* __restrict__ skip,
                                               const _Float16* __restrict__ Wf_t,
                                               const _Float16* __restrict__ Wg_t,
                                               const _Float16* __restrict__ Wpf_t,
                                               const _Float16* __restrict__ Wpg_t,
                                               const _Float16* __restrict__ Wr_t,
                                               const _Float16* __restrict__ Ws_t,
                                               const float* __restrict__ bf,
                                               const float* __restrict__ bg,
                                               int d, int Tcur, int cum) {
  int b = blockIdx.y;
  int wave = threadIdx.x >> 5, lane = threadIdx.x & 31;
  int half = lane >> 4, lc = lane & 15;
  int Tz = Tcur - d;
  int t0 = blockIdx.x * 64 + wave * 16;

  __shared__ _Float16 zt[4][16 * 128];
  __shared__ _Float16 rt[4][16 * 128];

  const _Float16* sigb = sig_in + (size_t)b * TPAD * 128;
  const _Float16* condb = cond + (size_t)b * TPAD * 256;

  Acc aF[8], aG[8];
#pragma unroll
  for (int nt = 0; nt < 8; nt++) {
    float vF = bf[nt * 16 + lc], vG = bg[nt * 16 + lc];
    for (int r = 0; r < 8; r++) { aF[nt].f[r] = vF; aG[nt].f[r] = vG; }
  }
  // dilated 2-tap conv: taps at t and t+d
#pragma unroll
  for (int tap = 0; tap < 2; tap++) {
    int roff = t0 + tap * d;
#pragma unroll
    for (int kt = 0; kt < 4; kt++) {
      v16h a = load_fragA(sigb, roff, 128, kt * 32);
#pragma unroll
      for (int nt = 0; nt < 8; nt++) {
        aF[nt].v = wmma16(a, load_fragB(Wf_t + (size_t)tap * 128 * 128, nt * 16, 128, kt * 32), aF[nt].v);
        aG[nt].v = wmma16(a, load_fragB(Wg_t + (size_t)tap * 128 * 128, nt * 16, 128, kt * 32), aG[nt].v);
      }
    }
  }
  // cond 1x1 (K=256)
#pragma unroll
  for (int kt = 0; kt < 8; kt++) {
    v16h a = load_fragA(condb, cum + t0, 256, kt * 32);
#pragma unroll
    for (int nt = 0; nt < 8; nt++) {
      aF[nt].v = wmma16(a, load_fragB(Wpf_t, nt * 16, 256, kt * 32), aF[nt].v);
      aG[nt].v = wmma16(a, load_fragB(Wpg_t, nt * 16, 256, kt * 32), aG[nt].v);
    }
  }
  // z = tanh(F) * sigmoid(G) -> LDS (16 x 128 f16, row = time)
  _Float16* zme = zt[wave];
#pragma unroll
  for (int nt = 0; nt < 8; nt++) {
    int n = nt * 16 + lc;
    for (int r = 0; r < 8; r++) {
      float g = aG[nt].f[r];
      float z = tanhf(aF[nt].f[r]) * (1.0f / (1.0f + __expf(-g)));
      zme[(half * 8 + r) * 128 + n] = (_Float16)z;
    }
  }
  __syncthreads();
  v16h za[4];
#pragma unroll
  for (int kt = 0; kt < 4; kt++) za[kt] = load_fragA_lds(zme, 128, kt * 32);

  // skip (N=256) accumulated into channel-major (B,256,T_OUT)
  float* skipb = skip + (size_t)b * 256 * T_OUT;
  int shift = TOTAL_D - cum;  // t_out = t_local - shift
#pragma unroll
  for (int nt = 0; nt < 16; nt++) {
    Acc acc;
    for (int r = 0; r < 8; r++) acc.f[r] = 0.f;
#pragma unroll
    for (int kt = 0; kt < 4; kt++)
      acc.v = wmma16(za[kt], load_fragB(Ws_t, nt * 16, 128, kt * 32), acc.v);
    int n = nt * 16 + lc;
    int tl = t0 + half * 8;
    float* sp = skipb + (size_t)n * T_OUT;
    for (int r = 0; r < 8; r++) {
      int to = tl + r - shift;
      if (to >= 0 && (tl + r) < Tz) sp[to] += acc.f[r];
    }
  }
  // res (N=128) = z @ Wr + sig[t+d] -> LDS -> contiguous store
  _Float16* rme = rt[wave];
#pragma unroll
  for (int nt = 0; nt < 8; nt++) {
    Acc acc;
    for (int r = 0; r < 8; r++) acc.f[r] = 0.f;
#pragma unroll
    for (int kt = 0; kt < 4; kt++)
      acc.v = wmma16(za[kt], load_fragB(Wr_t, nt * 16, 128, kt * 32), acc.v);
    int n = nt * 16 + lc;
    for (int r = 0; r < 8; r++) {
      int tl = t0 + half * 8 + r;
      float s = (float)sigb[(size_t)(tl + d) * 128 + n];
      rme[(half * 8 + r) * 128 + n] = (_Float16)(acc.f[r] + s);
    }
  }
  __syncthreads();
  {
    int r = lane & 15, hh = lane >> 4;
    int tl = t0 + r;
    if (tl < Tz) {
      const v8h* src = (const v8h*)(rme + r * 128 + hh * 64);
      v8h* dst = (v8h*)(sig_out + ((size_t)b * TPAD + tl) * 128 + hh * 64);
#pragma unroll
      for (int i = 0; i < 8; i++) dst[i] = src[i];
    }
  }
}

// post1: h = relu(relu(skip) @ W_post1^T + b) stored (B,TOPAD,256) f16
__global__ __launch_bounds__(128) void k_post1(const float* __restrict__ skip,
                                               const _Float16* __restrict__ Wt,
                                               const float* __restrict__ bias,
                                               _Float16* __restrict__ h) {
  int b = blockIdx.y;
  int wave = threadIdx.x >> 5, lane = threadIdx.x & 31;
  int half = lane >> 4, lc = lane & 15;
  int t0 = blockIdx.x * 64 + wave * 16;
  __shared__ _Float16 st[4][16 * 256];
  const float* skipb = skip + (size_t)b * 256 * T_OUT;
  Acc acc[16];
#pragma unroll
  for (int nt = 0; nt < 16; nt++) {
    float v = bias[nt * 16 + lc];
    for (int r = 0; r < 8; r++) acc[nt].f[r] = v;
  }
  int row = t0 + lc;
  if (row >= T_OUT) row = T_OUT - 1;  // clamp pad rows (discarded at store)
  for (int kt = 0; kt < 8; kt++) {
    Frag a;
    int k0 = kt * 32;
#pragma unroll
    for (int j = 0; j < 8; j++) {
      a.h[j]     = (_Float16)fmaxf(skipb[(size_t)(k0 + half * 8 + j) * T_OUT + row], 0.f);
      a.h[8 + j] = (_Float16)fmaxf(skipb[(size_t)(k0 + 16 + half * 8 + j) * T_OUT + row], 0.f);
    }
#pragma unroll
    for (int nt = 0; nt < 16; nt++)
      acc[nt].v = wmma16(a.v, load_fragB(Wt, nt * 16, 256, k0), acc[nt].v);
  }
  _Float16* tile = st[wave];
#pragma unroll
  for (int nt = 0; nt < 16; nt++) {
    int n = nt * 16 + lc;
    for (int r = 0; r < 8; r++)
      tile[(half * 8 + r) * 256 + n] = (_Float16)fmaxf(acc[nt].f[r], 0.f);
  }
  __syncthreads();
  int r = lane & 15, hh = lane >> 4;
  int tl = t0 + r;
  if (tl < T_OUT) {
    const v8h* src = (const v8h*)(tile + r * 256 + hh * 128);
    v8h* dst = (v8h*)(h + ((size_t)b * TOPAD + tl) * 256 + hh * 128);
#pragma unroll
    for (int i = 0; i < 16; i++) dst[i] = src[i];
  }
}

// post2: out (B,256,T_OUT) f32 = h @ W_post2^T + b  (D-fragment -> contiguous t stores)
__global__ __launch_bounds__(128) void k_post2(const _Float16* __restrict__ h,
                                               const _Float16* __restrict__ Wt,
                                               const float* __restrict__ bias,
                                               float* __restrict__ out) {
  int b = blockIdx.y;
  int wave = threadIdx.x >> 5, lane = threadIdx.x & 31;
  int half = lane >> 4, lc = lane & 15;
  int t0 = blockIdx.x * 64 + wave * 16;
  const _Float16* hb = h + (size_t)b * TOPAD * 256;
  Acc acc[16];
#pragma unroll
  for (int nt = 0; nt < 16; nt++) {
    float v = bias[nt * 16 + lc];
    for (int r = 0; r < 8; r++) acc[nt].f[r] = v;
  }
  for (int kt = 0; kt < 8; kt++) {
    v16h a = load_fragA(hb, t0, 256, kt * 32);
#pragma unroll
    for (int nt = 0; nt < 16; nt++)
      acc[nt].v = wmma16(a, load_fragB(Wt, nt * 16, 256, kt * 32), acc[nt].v);
  }
#pragma unroll
  for (int nt = 0; nt < 16; nt++) {
    int n = nt * 16 + lc;
    int tl = t0 + half * 8;
    float* op = out + ((size_t)b * 256 + n) * T_OUT + tl;
    for (int r = 0; r < 8; r++)
      if (tl + r < T_OUT) op[r] = acc[nt].f[r];
  }
}

// ---------------- host launcher ----------------

extern "C" void kernel_launch(void* const* d_in, const int* in_sizes, int n_in,
                              void* d_out, int out_size, void* d_ws, size_t ws_size,
                              hipStream_t stream) {
  (void)in_sizes; (void)n_in; (void)out_size; (void)ws_size;
  const float* wav      = (const float*)d_in[0];
  const float* lc_sp    = (const float*)d_in[1];
  const int*   spk      = (const int*)d_in[2];
  const int*   jit      = (const int*)d_in[3];
  const float* W_lc     = (const float*)d_in[4];
  const float* b_lc     = (const float*)d_in[5];
  const float* W_up[4]  = {(const float*)d_in[6], (const float*)d_in[8],
                           (const float*)d_in[10], (const float*)d_in[12]};
  const float* b_up[4]  = {(const float*)d_in[7], (const float*)d_in[9],
                           (const float*)d_in[11], (const float*)d_in[13]};
  const float* W_spk    = (const float*)d_in[14];
  const float* b_spk    = (const float*)d_in[15];
  const float* W_base   = (const float*)d_in[16];
  const float* b_base   = (const float*)d_in[17];
  const float* Wf       = (const float*)d_in[18];
  const float* bf       = (const float*)d_in[19];
  const float* Wg       = (const float*)d_in[20];
  const float* bg       = (const float*)d_in[21];
  const float* Wpf      = (const float*)d_in[22];
  const float* Wpg      = (const float*)d_in[23];
  const float* Wr       = (const float*)d_in[24];
  const float* Ws       = (const float*)d_in[25];
  const float* W_post1  = (const float*)d_in[26];
  const float* b_post1  = (const float*)d_in[27];
  const float* W_post2  = (const float*)d_in[28];
  const float* b_post2  = (const float*)d_in[29];

  char* ws = (char*)d_ws;
  size_t cur = 0;
  auto alloc = [&](size_t bytes) -> char* {
    char* p = ws + cur;
    cur += (bytes + 255) & ~(size_t)255;
    return p;
  };

  _Float16* wavT   = (_Float16*)alloc((size_t)B_ * TPAD * 256 * 2);
  _Float16* sigA   = (_Float16*)alloc((size_t)B_ * TPAD * 128 * 2);
  _Float16* sigB   = (_Float16*)alloc((size_t)B_ * TPAD * 128 * 2);
  _Float16* cond   = (_Float16*)alloc((size_t)B_ * TPAD * 256 * 2);
  float*    skip   = (float*)   alloc((size_t)B_ * 256 * T_OUT * 4);
  _Float16* hbuf   = (_Float16*)alloc((size_t)B_ * TOPAD * 256 * 2);
  _Float16* Wbase_t= (_Float16*)alloc((size_t)128 * 256 * 2);
  _Float16* Wf_t   = (_Float16*)alloc((size_t)NLAY * 2 * 128 * 128 * 2);
  _Float16* Wg_t   = (_Float16*)alloc((size_t)NLAY * 2 * 128 * 128 * 2);
  _Float16* Wpf_t  = (_Float16*)alloc((size_t)NLAY * 128 * 256 * 2);
  _Float16* Wpg_t  = (_Float16*)alloc((size_t)NLAY * 128 * 256 * 2);
  _Float16* Wr_t   = (_Float16*)alloc((size_t)NLAY * 128 * 128 * 2);
  _Float16* Ws_t   = (_Float16*)alloc((size_t)NLAY * 256 * 128 * 2);
  _Float16* Wp1_t  = (_Float16*)alloc((size_t)256 * 256 * 2);
  _Float16* Wp2_t  = (_Float16*)alloc((size_t)256 * 256 * 2);
  float*    lc0    = (float*)alloc((size_t)B_ * 128 * 40 * 4);
  float*    u0     = (float*)alloc((size_t)B_ * 128 * 156 * 4);
  float*    u1     = (float*)alloc((size_t)B_ * 128 * 620 * 4);
  float*    u2     = (float*)alloc((size_t)B_ * 128 * 2476 * 4);
  float*    u3     = (float*)alloc((size_t)B_ * 128 * T_U3 * 4);

  auto nblk = [](long n) { return (unsigned)((n + 255) / 256); };

  // weight conversion
  k_w2t<<<nblk((long)128 * 256), 256, 0, stream>>>(W_base, Wbase_t, 1, 128, 256, 1);
  k_w2t<<<nblk((long)NLAY * 128 * 128 * 2), 256, 0, stream>>>(Wf, Wf_t, NLAY, 128, 128, 2);
  k_w2t<<<nblk((long)NLAY * 128 * 128 * 2), 256, 0, stream>>>(Wg, Wg_t, NLAY, 128, 128, 2);
  k_w2t<<<nblk((long)NLAY * 128 * 256), 256, 0, stream>>>(Wpf, Wpf_t, NLAY, 128, 256, 1);
  k_w2t<<<nblk((long)NLAY * 128 * 256), 256, 0, stream>>>(Wpg, Wpg_t, NLAY, 128, 256, 1);
  k_w2t<<<nblk((long)NLAY * 128 * 128), 256, 0, stream>>>(Wr, Wr_t, NLAY, 128, 128, 1);
  k_w2t<<<nblk((long)NLAY * 256 * 128), 256, 0, stream>>>(Ws, Ws_t, NLAY, 256, 128, 1);
  k_w2t<<<nblk((long)256 * 256), 256, 0, stream>>>(W_post1, Wp1_t, 1, 256, 256, 1);
  k_w2t<<<nblk((long)256 * 256), 256, 0, stream>>>(W_post2, Wp2_t, 1, 256, 256, 1);

  // activations prep
  k_wavT<<<nblk((long)B_ * T_WAV * 256), 256, 0, stream>>>(wav, wavT);
  k_zero_f32<<<nblk((long)B_ * 256 * T_OUT), 256, 0, stream>>>(skip, (long)B_ * 256 * T_OUT);

  // local-conditioning chain
  k_lcconv<<<nblk((long)B_ * 128 * 40), 256, 0, stream>>>(lc_sp, jit, W_lc, b_lc, lc0);
  k_upconv<<<nblk((long)B_ * 128 * 156), 256, 0, stream>>>(lc0, W_up[0], b_up[0], u0, 40, 156, 4, 8);
  k_upconv<<<nblk((long)B_ * 128 * 620), 256, 0, stream>>>(u0, W_up[1], b_up[1], u1, 156, 620, 4, 8);
  k_upconv<<<nblk((long)B_ * 128 * 2476), 256, 0, stream>>>(u1, W_up[2], b_up[2], u2, 620, 2476, 4, 8);
  k_upconv<<<nblk((long)B_ * 128 * T_U3), 256, 0, stream>>>(u2, W_up[3], b_up[3], u3, 2476, T_U3, 5, 10);
  k_cond<<<nblk((long)B_ * T_WAV * 256), 256, 0, stream>>>(u3, W_spk, b_spk, spk, cond);

  // base 1x1: sig = wav_onehot @ W_base + b_base
  {
    dim3 g((T_WAV + 63) / 64, B_);
    k_gemm_n128<<<g, 128, 0, stream>>>(wavT, Wbase_t, b_base, sigA, T_WAV, 256);
  }

  // residual stack
  static const int dils[NLAY] = {1, 2, 4, 8, 16, 32, 64, 128, 256, 512,
                                 1, 2, 4, 8, 16, 32, 64, 128, 256, 512};
  _Float16* sin = sigA;
  _Float16* sout = sigB;
  int Tcur = T_WAV, cum = 0;
  for (int i = 0; i < NLAY; i++) {
    int d = dils[i];
    cum += d;
    int Tz = Tcur - d;
    dim3 g((Tz + 63) / 64, B_);
    k_layer<<<g, 128, 0, stream>>>(sin, sout, cond, skip,
                                   Wf_t + (size_t)i * 2 * 128 * 128,
                                   Wg_t + (size_t)i * 2 * 128 * 128,
                                   Wpf_t + (size_t)i * 128 * 256,
                                   Wpg_t + (size_t)i * 128 * 256,
                                   Wr_t + (size_t)i * 128 * 128,
                                   Ws_t + (size_t)i * 256 * 128,
                                   bf + (size_t)i * 128, bg + (size_t)i * 128,
                                   d, Tcur, cum);
    _Float16* tmp = sin; sin = sout; sout = tmp;
    Tcur = Tz;
  }

  // post network
  {
    dim3 g((T_OUT + 63) / 64, B_);
    k_post1<<<g, 128, 0, stream>>>(skip, Wp1_t, b_post1, hbuf);
    k_post2<<<g, 128, 0, stream>>>(hbuf, Wp2_t, b_post2, (float*)d_out);
  }
}
